// RNN_19602230739611
// MI455X (gfx1250) — matrix operand order repaired and checked by reading, not compiled, not run on hardware
//
#include <hip/hip_runtime.h>
#include <hip/hip_bf16.h>
#include <math.h>

// ---------------------------------------------------------------------------
// Elman RNN fused kernel for gfx1250 (MI455X).
//   h_t = tanh(x_t Wih^T + b_ih + h_{t-1} Whh^T + b_hh), t = 0..1023
//   out = log_softmax(h_T Who^T + b_ho)
// bf16 WMMA (v_wmma_f32_16x16x32_bf16) with fp32 accumulate.
// Grid: 64 workgroups x 256 threads (8 waves). Each WG owns 16 batch rows;
// each wave owns a 16-column slice of H=128. Weights staged to LDS once,
// hoisted into persistent B fragments. h shared via LDS each step.
// tanh via native V_TANH_F32 (or branch-free exp2/rcp fallback) to keep the
// serial per-step critical path short.
// ---------------------------------------------------------------------------

typedef __attribute__((ext_vector_type(16))) __bf16 v16bf;
typedef __attribute__((ext_vector_type(8)))  __bf16 v8bf;
typedef __attribute__((ext_vector_type(8)))  float  v8f;

#define RNN_SEQ   1024
#define RNN_B     1024
#define RNN_IN    57
#define RNN_H     128
#define RNN_OUT   18
#define RNN_KX    64      // IN padded to a multiple of 32 (WMMA K)
#define RNN_TM    16      // batch rows per workgroup

__device__ __forceinline__ float fast_tanh(float x) {
#if __has_builtin(__builtin_amdgcn_tanhf)
  return __builtin_amdgcn_tanhf(x);            // v_tanh_f32 (TRANS32)
#else
  // tanh(x) = sign(x) * (1 - 2t/(1+t)),  t = exp(-2|x|) = 2^(-2|x|*log2 e)
  float ax = __builtin_fabsf(x);
  float t  = __builtin_amdgcn_exp2f(ax * -2.885390081777927f);  // v_exp_f32
  float r  = __builtin_amdgcn_rcpf(1.0f + t);                   // v_rcp_f32
  float y  = 1.0f - (2.0f * t) * r;
  return __builtin_copysignf(y, x);
#endif
}

__device__ __forceinline__ v16bf cat8(v8bf a, v8bf b) {
  v16bf r;
#pragma unroll
  for (int i = 0; i < 8; ++i) { r[i] = a[i]; r[i + 8] = b[i]; }
  return r;
}

// A-matrix fragment, 16x32 bf16 (ISA 7.12.2): lanes 0-15 hold M=lane,
// elems 0..7 = K kbase..kbase+7, elems 8..15 = K kbase+16..kbase+23;
// lanes 16-31 shift both K runs by +8.
__device__ __forceinline__ v16bf ld_afrag(const __bf16* row, int kbase, int hi) {
  v8bf lo = *(const v8bf*)(row + kbase + (hi ? 8 : 0));
  v8bf up = *(const v8bf*)(row + kbase + 16 + (hi ? 8 : 0));
  return cat8(lo, up);
}

// B-matrix fragment, 32x16 bf16: lane holds column N=lane%16, elems 0..7 =
// K kbase..kbase+7, elems 8..15 = K kbase+16..kbase+23 (+8 for lanes 16-31).
// `col` points at a column stored contiguously in K.
__device__ __forceinline__ v16bf ld_bfrag(const __bf16* col, int kbase, int hi) {
  int klo = kbase + (hi ? 8 : 0);
  v8bf a = *(const v8bf*)(col + klo);
  v8bf b = *(const v8bf*)(col + klo + 16);
  return cat8(a, b);
}

__global__ __launch_bounds__(256) void rnn_fused_kernel(
    const float* __restrict__ x,    // [SEQ, B, IN]
    const float* __restrict__ Wih,  // [H, IN]
    const float* __restrict__ bih,  // [H]
    const float* __restrict__ Whh,  // [H, H]
    const float* __restrict__ bhh,  // [H]
    const float* __restrict__ Who,  // [OUT, H]
    const float* __restrict__ bho,  // [OUT]
    float* __restrict__ out)        // [B*OUT] ++ [B*H]
{
  // LDS: weights in B-fragment-friendly (column-major => source row-major)
  // bf16 layout, plus per-step x tile, bf16 h tile, fp32 h_T, fused bias.
  __shared__ __align__(32) __bf16 sWih[RNN_H * RNN_KX];  // [n][k], k padded
  __shared__ __align__(32) __bf16 sWhh[RNN_H * RNN_H];   // [n][k]
  __shared__ __align__(32) __bf16 sX[RNN_TM * RNN_KX];   // [m][k]
  __shared__ __align__(32) __bf16 sH[RNN_TM * RNN_H];    // [m][k]
  __shared__ float sHf[RNN_TM * RNN_H];
  __shared__ float sBias[RNN_H];

  const int tid  = threadIdx.x;
  const int wave = tid >> 5;        // 0..7 -> which 16-col slice of H
  const int lane = tid & 31;
  const int hi   = lane >> 4;       // upper half-wave
  const int rowA = lane & 15;       // A-fragment row
  const int ncol = wave * 16 + (lane & 15);  // this lane's H column
  const int b0   = blockIdx.x * RNN_TM;      // batch tile base

  // ---- stage weights (fp32 -> bf16), bias, zero h0 -------------------------
  for (int e = tid; e < RNN_H * RNN_KX; e += 256) {
    int n = e >> 6, k = e & 63;
    sWih[e] = (__bf16)((k < RNN_IN) ? Wih[n * RNN_IN + k] : 0.f);
  }
  for (int e = tid; e < RNN_H * RNN_H; e += 256) {
    int n = e >> 7, k = e & 127;
    sWhh[e] = (__bf16)Whh[n * RNN_H + k];
  }
  if (tid < RNN_H) sBias[tid] = bih[tid] + bhh[tid];
  for (int e = tid; e < RNN_TM * RNN_H; e += 256) sH[e] = (__bf16)0.f;
  __syncthreads();

  // ---- persistent B fragments for this wave's column slice -----------------
  v16bf bWih0 = ld_bfrag(&sWih[ncol * RNN_KX], 0,  hi);
  v16bf bWih1 = ld_bfrag(&sWih[ncol * RNN_KX], 32, hi);
  v16bf bWhh0 = ld_bfrag(&sWhh[ncol * RNN_H],  0,  hi);
  v16bf bWhh1 = ld_bfrag(&sWhh[ncol * RNN_H],  32, hi);
  v16bf bWhh2 = ld_bfrag(&sWhh[ncol * RNN_H],  64, hi);
  v16bf bWhh3 = ld_bfrag(&sWhh[ncol * RNN_H],  96, hi);

  // bias-splatted accumulator init, hoisted out of the time loop
  const float bias = sBias[ncol];
  v8f cbias;
#pragma unroll
  for (int j = 0; j < 8; ++j) cbias[j] = bias;

  v8f d;  // fp32 h slice for this lane (8 rows x 1 col), survives the loop
#pragma unroll
  for (int j = 0; j < 8; ++j) d[j] = 0.f;

#pragma unroll 1
  for (int t = 0; t < RNN_SEQ; ++t) {
    // stage x_t tile: fp32 global -> bf16 LDS, zero-padded to KX columns
    const float* xt = x + ((size_t)t * RNN_B + b0) * RNN_IN;
#pragma unroll
    for (int s = tid; s < RNN_TM * RNN_KX; s += 256) {
      int r = s >> 6, k = s & 63;
      sX[s] = (__bf16)((k < RNN_IN) ? xt[r * RNN_IN + k] : 0.f);
    }
    // prefetch next timestep's x tile (16*57*4 = 3648 B) into cache
    if (t + 1 < RNN_SEQ) {
      const float* xn = x + ((size_t)(t + 1) * RNN_B + b0) * RNN_IN;
      if (tid < 29) __builtin_prefetch(xn + tid * 32, 0, 1);
    }
    __syncthreads();  // x_t staged; previous step's sH writes visible

    // A fragments from LDS (ds_load_b128 pairs)
    v16bf ax0 = ld_afrag(&sX[rowA * RNN_KX], 0,  hi);
    v16bf ax1 = ld_afrag(&sX[rowA * RNN_KX], 32, hi);
    v16bf ah0 = ld_afrag(&sH[rowA * RNN_H],  0,  hi);
    v16bf ah1 = ld_afrag(&sH[rowA * RNN_H],  32, hi);
    v16bf ah2 = ld_afrag(&sH[rowA * RNN_H],  64, hi);
    v16bf ah3 = ld_afrag(&sH[rowA * RNN_H],  96, hi);
    __syncthreads();  // all fragment reads done; sX/sH may be overwritten

    // accumulate: bias + x_t Wih^T + h_{t-1} Whh^T  (6 WMMAs)
    v8f c = cbias;
    c = __builtin_amdgcn_wmma_f32_16x16x32_bf16(false, ax0, false, bWih0, (short)0, c, false, false);
    c = __builtin_amdgcn_wmma_f32_16x16x32_bf16(false, ax1, false, bWih1, (short)0, c, false, false);
    c = __builtin_amdgcn_wmma_f32_16x16x32_bf16(false, ah0, false, bWhh0, (short)0, c, false, false);
    c = __builtin_amdgcn_wmma_f32_16x16x32_bf16(false, ah1, false, bWhh1, (short)0, c, false, false);
    c = __builtin_amdgcn_wmma_f32_16x16x32_bf16(false, ah2, false, bWhh2, (short)0, c, false, false);
    c = __builtin_amdgcn_wmma_f32_16x16x32_bf16(false, ah3, false, bWhh3, (short)0, c, false, false);

    // branch-free tanh on the serial critical path
#pragma unroll
    for (int j = 0; j < 8; ++j) d[j] = fast_tanh(c[j]);

    // publish bf16 h slice: C/D layout -> row = j + 8*hi, col = ncol
#pragma unroll
    for (int j = 0; j < 8; ++j) {
      int mrow = j + (hi ? 8 : 0);
      sH[mrow * RNN_H + ncol] = (__bf16)d[j];
    }
  }

  // ---- epilogue: h_T out (fp32) + h2o + log_softmax ------------------------
  float* hT = out + (size_t)RNN_B * RNN_OUT;
#pragma unroll
  for (int j = 0; j < 8; ++j) {
    int mrow = j + (hi ? 8 : 0);
    sHf[mrow * RNN_H + ncol] = d[j];
    hT[(size_t)(b0 + mrow) * RNN_H + ncol] = d[j];
  }
  __syncthreads();

  if (tid < RNN_TM) {  // one thread per batch row; 18x128 MACs, negligible
    const float* hr = &sHf[tid * RNN_H];
    float logits[RNN_OUT];
#pragma unroll 1
    for (int o = 0; o < RNN_OUT; ++o) {
      float s = bho[o];
      const float* wr = Who + o * RNN_H;
#pragma unroll 4
      for (int k = 0; k < RNN_H; ++k) s = fmaf(hr[k], wr[k], s);
      logits[o] = s;
    }
    float m = logits[0];
#pragma unroll
    for (int o = 1; o < RNN_OUT; ++o) m = fmaxf(m, logits[o]);
    float sum = 0.f;
#pragma unroll
    for (int o = 0; o < RNN_OUT; ++o) sum += expf(logits[o] - m);
    float lse = m + logf(sum);
    float* orow = out + (size_t)(b0 + tid) * RNN_OUT;
#pragma unroll
    for (int o = 0; o < RNN_OUT; ++o) orow[o] = logits[o] - lse;
  }
}

extern "C" void kernel_launch(void* const* d_in, const int* in_sizes, int n_in,
                              void* d_out, int out_size, void* d_ws, size_t ws_size,
                              hipStream_t stream) {
  (void)in_sizes; (void)n_in; (void)out_size; (void)d_ws; (void)ws_size;
  const float* x   = (const float*)d_in[0];
  const float* Wih = (const float*)d_in[1];
  const float* bih = (const float*)d_in[2];
  const float* Whh = (const float*)d_in[3];
  const float* bhh = (const float*)d_in[4];
  const float* Who = (const float*)d_in[5];
  const float* bho = (const float*)d_in[6];
  float* out = (float*)d_out;

  dim3 grid(RNN_B / RNN_TM);   // 64 workgroups
  dim3 block(256);             // 8 wave32 per workgroup
  rnn_fused_kernel<<<grid, block, 0, stream>>>(x, Wih, bih, Whh, bhh, Who, bho, out);
}
